// LIFDenseLayer_31645319036935
// MI455X (gfx1250) — 8-bit matrix operand order repaired and checked by reading, compiled, not run on hardware
//
#include <hip/hip_runtime.h>


typedef int v8i __attribute__((ext_vector_type(8)));

#define ALPHA 0.8187307530779818f
#define THRESH 0.4f

#define WT_STRIDE 528
#define A_OFF     (256 * WT_STRIDE)
#define A_STRIDE  72
#define SMEM_BYTES (A_OFF + 64 * A_STRIDE)

#if defined(__has_builtin)
#  if __has_builtin(__builtin_amdgcn_global_load_async_to_lds_b128) && \
      __has_builtin(__builtin_amdgcn_s_wait_asynccnt)
#    define USE_ASYNC_LDS 1
#  endif
#endif

#ifdef USE_ASYNC_LDS
typedef int async_v4i __attribute__((vector_size(16)));
typedef __attribute__((address_space(1))) async_v4i* g1p_v4i;
typedef __attribute__((address_space(3))) async_v4i* l3p_v4i;
#endif

__device__ __forceinline__ int q8i(float x) {
    x = fminf(fmaxf(x, -0.9921875f), 0.9921875f);
    return (int)rintf(x * 128.0f);
}

__global__ __launch_bounds__(256) void quant_wt_kernel(
    const float* __restrict__ W, signed char* __restrict__ Wt)
{
    int idx = blockIdx.x * 256 + threadIdx.x;
    int k = idx >> 9;
    int o = idx & 511;
    Wt[o * 512 + k] = (signed char)q8i(W[idx]);
}

__global__ __launch_bounds__(256) void lif_gemm_kernel(
    const float* __restrict__ P, const float* __restrict__ Q,
    const float* __restrict__ S, const signed char* __restrict__ Wt,
    const float* __restrict__ bias, float* __restrict__ out)
{
    extern __shared__ unsigned char smem[];
    unsigned char* wt_lds = smem;
    unsigned char* a_lds  = smem + A_OFF;

    const int tid  = threadIdx.x;
    const int lane = tid & 31;
    const int wave = tid >> 5;
    const int rowGroup = wave >> 1;
    const int colGroup = wave & 1;
    const int rowBase_g = blockIdx.y * 64;
    const int colBase_g = blockIdx.x * 256;

#ifdef USE_ASYNC_LDS
    {
        unsigned char* srcb = (unsigned char*)(Wt + colBase_g * 512);
        for (int e = tid; e < 8192; e += 256) {
            int c  = e >> 5;
            int kw = e & 31;
            __builtin_amdgcn_global_load_async_to_lds_b128(
                (g1p_v4i)(srcb + e * 16),
                (l3p_v4i)(wt_lds + c * WT_STRIDE + kw * 16),
                0, 0);
        }
        __builtin_amdgcn_s_wait_asynccnt(0);
    }
#else
    {
        const uint4* src = (const uint4*)(Wt + colBase_g * 512);
        for (int e = tid; e < 8192; e += 256) {
            int c  = e >> 5;
            int kw = e & 31;
            *(uint4*)(wt_lds + c * WT_STRIDE + kw * 16) = src[e];
        }
    }
#endif

    v8i acc[8] = {};

    const float4* P4 = (const float4*)P;
    const float4* Q4 = (const float4*)Q;
    const int k4 = tid & 15;
    const int rb = tid >> 4;

    for (int kc = 0; kc < 512; kc += 64) {
        __syncthreads();

        #pragma unroll
        for (int rr = 0; rr < 4; ++rr) {
            int row = rb + rr * 16;
            int gi  = (((rowBase_g + row) << 9) + kc + (k4 << 2)) >> 2;
            float4 p = P4[gi];
            float4 q = Q4[gi];
            int q0 = q8i(ALPHA * p.x + q.x);
            int q1 = q8i(ALPHA * p.y + q.y);
            int q2 = q8i(ALPHA * p.z + q.z);
            int q3 = q8i(ALPHA * p.w + q.w);
            unsigned int packed = (unsigned)(q0 & 255) | ((unsigned)(q1 & 255) << 8) |
                                  ((unsigned)(q2 & 255) << 16) | ((unsigned)(q3 & 255) << 24);
            *(unsigned int*)(a_lds + row * A_STRIDE + (k4 << 2)) = packed;
        }
        __syncthreads();

        const int m  = lane & 15;
        const int hi = lane >> 4;
        const unsigned char* abase = a_lds + (rowGroup * 16 + m) * A_STRIDE + hi * 16;
        unsigned long long a01 = *(const unsigned long long*)(abase +  0);
        unsigned long long a23 = *(const unsigned long long*)(abase +  8);
        unsigned long long a45 = *(const unsigned long long*)(abase + 32);
        unsigned long long a67 = *(const unsigned long long*)(abase + 40);
        v8i Af;
        Af[0] = (int)a01; Af[1] = (int)(a01 >> 32);
        Af[2] = (int)a23; Af[3] = (int)(a23 >> 32);
        Af[4] = (int)a45; Af[5] = (int)(a45 >> 32);
        Af[6] = (int)a67; Af[7] = (int)(a67 >> 32);

        const int koff = hi * 16;
        #pragma unroll
        for (int t = 0; t < 8; ++t) {
            int c_local = colGroup * 128 + t * 16 + (lane & 15);
            const unsigned char* bbase = wt_lds + c_local * WT_STRIDE + kc + koff;
            int4 b0 = *(const int4*)(bbase);
            int4 b1 = *(const int4*)(bbase + 32);
            v8i Bf;
            Bf[0] = b0.x; Bf[1] = b0.y; Bf[2] = b0.z; Bf[3] = b0.w;
            Bf[4] = b1.x; Bf[5] = b1.y; Bf[6] = b1.z; Bf[7] = b1.w;
            acc[t] = __builtin_amdgcn_wmma_i32_16x16x64_iu8(
                true, Af, true, Bf, acc[t], false, false);
        }
    }

    const int n_lane = lane & 15;
    const int m_off  = (lane >> 4) * 8;
    #pragma unroll
    for (int t = 0; t < 8; ++t) {
        int n = colBase_g + colGroup * 128 + t * 16 + n_lane;
        float bq = (float)q8i(bias[n]) * (1.0f / 128.0f);
        int m0 = rowBase_g + rowGroup * 16 + m_off;
        #pragma unroll
        for (int r = 0; r < 8; ++r) {
            int idx = (m0 + r) * 512 + n;
            float u = (float)acc[t][r] * (1.0f / 16384.0f) + bq - S[idx];
            out[idx] = (u > THRESH) ? 1.0f : 0.0f;
        }
    }
}

extern "C" void kernel_launch(void* const* d_in, const int* in_sizes, int n_in,
                              void* d_out, int out_size, void* d_ws, size_t ws_size,
                              hipStream_t stream) {
    const float* P    = (const float*)d_in[1];
    const float* Q    = (const float*)d_in[2];
    const float* S    = (const float*)d_in[4];
    const float* W    = (const float*)d_in[5];
    const float* bias = (const float*)d_in[6];
    float* out        = (float*)d_out;
    signed char* wt_q = (signed char*)d_ws;

    quant_wt_kernel<<<dim3(1024), dim3(256), 0, stream>>>(W, wt_q);
    lif_gemm_kernel<<<dim3(2, 512), dim3(256), SMEM_BYTES, stream>>>(
        P, Q, S, wt_q, bias, out);
}
